// SynthesisLayer_68178310856802
// MI455X (gfx1250) — matrix operand order repaired and checked by reading, not compiled, not run on hardware
//
#include <hip/hip_runtime.h>
#include <hip/hip_bf16.h>
#include <math.h>

typedef _Float16 v16h __attribute__((ext_vector_type(16)));
typedef _Float16 v8h  __attribute__((ext_vector_type(8)));
typedef _Float16 v2h  __attribute__((ext_vector_type(2)));
typedef float    v8f  __attribute__((ext_vector_type(8)));
typedef float    v2f  __attribute__((ext_vector_type(2)));

#define Hdim 256
#define Wdim 256
#define Cin  128
#define Cout 128
#define WDIM 512
#define PXS  136            // padded pixel stride in halfs (128 + 8 pad)
#define ROWS (130 * PXS)    // halfs per halo row
#define XS_HALFS (3 * ROWS)

// ---------------- prep 1: s = z @ (Aw/sqrt(512)) + b ----------------
__global__ void prep_affine(const float* __restrict__ dlat,
                            const float* __restrict__ aw,
                            const float* __restrict__ ab,
                            const int* __restrict__ layer_idx,
                            float* __restrict__ sBuf) {
  const int n = blockIdx.x;           // 8 blocks
  const int c = threadIdx.x;          // 128 threads
  const int li = layer_idx[0];
  const float* z = dlat + ((size_t)n * 16 + li) * WDIM;
  float acc = 0.f;
  for (int k = 0; k < WDIM; ++k)
    acc += z[k] * aw[(size_t)k * Cin + c];
  sBuf[n * Cin + c] = acc * (1.0f / sqrtf((float)WDIM)) + ab[c];
}

// ---------------- prep 2: d[n,o] = rsqrt(sum_{t,i} (w*g)^2 * s^2 + 1e-8) ----------------
__global__ void prep_demod(const float* __restrict__ cw,
                           const float* __restrict__ sBuf,
                           float* __restrict__ dBuf) {
  __shared__ float s2[Cin];
  const int n = blockIdx.x;           // 8 blocks
  const int o = threadIdx.x;          // 128 threads
  {
    float sv = sBuf[n * Cin + o];
    s2[o] = sv * sv;
  }
  __syncthreads();
  const float g = 1.0f / sqrtf(9.0f * (float)Cin);
  float acc = 0.f;
  for (int t = 0; t < 9; ++t)
    for (int i = 0; i < Cin; ++i) {
      float wv = cw[((size_t)t * Cin + i) * Cout + o] * g;
      acc += wv * wv * s2[i];
    }
  dBuf[n * Cout + o] = __frsqrt_rn(acc + 1e-8f);
}

// ---------------- prep 3: wT[t][o][i] = conv_w[t][i][o] * g  (f16, B-fragment friendly) ----------------
__global__ void prep_weights(const float* __restrict__ cw,
                             _Float16* __restrict__ wT) {
  const int idx = blockIdx.x * blockDim.x + threadIdx.x;  // 9*128*128
  const int t = idx >> 14;
  const int rem = idx & 16383;
  const int o = rem >> 7;
  const int i = rem & 127;
  const float g = 1.0f / sqrtf(9.0f * (float)Cin);
  wT[idx] = (_Float16)(cw[((size_t)t * Cin + i) * Cout + o] * g);
}

// ---------------- main: implicit-GEMM modulated conv3x3 + fused epilogue ----------------
__global__ void __launch_bounds__(256)
synth_main(const float* __restrict__ x,
           const float* __restrict__ noise,
           const float* __restrict__ sBuf,
           const float* __restrict__ dBuf,
           const _Float16* __restrict__ wT,
           const float* __restrict__ bias,
           const float* __restrict__ nsPtr,
           float* __restrict__ out) {
  extern __shared__ __align__(16) char smem[];
  float* sS = (float*)smem;                       // 128
  float* dS = sS + Cin;                           // 128
  float* bS = dS + Cout;                          // 128
  _Float16* xs = (_Float16*)(smem + 3 * Cin * sizeof(float));

  const int w0 = blockIdx.x * 128;   // 0 or 128
  const int h  = blockIdx.y;         // 0..255
  const int n  = blockIdx.z;         // 0..7
  const int tid = threadIdx.x;

  if (tid < 128) {
    sS[tid] = sBuf[n * Cin + tid];
    dS[tid] = dBuf[n * Cout + tid];
    bS[tid] = bias[tid];
  }
  __syncthreads();

  // ---- stage modulated x halo (3 rows x 130 px x 128 ch) as f16 into LDS ----
  for (int i = tid; i < 3 * 130 * 64; i += 256) {
    const int r   = i / (130 * 64);
    const int rem = i - r * (130 * 64);
    const int p   = rem >> 6;        // 0..129
    const int c2  = rem & 63;        // channel pair
    const int hh = h + r - 1;
    const int ww = w0 + p - 1;
    v2f val = {0.f, 0.f};
    if ((unsigned)hh < (unsigned)Hdim && (unsigned)ww < (unsigned)Wdim) {
      val = *(const v2f*)(x + (((size_t)n * Hdim + hh) * Wdim + ww) * Cin + c2 * 2);
    }
    const int c = c2 * 2;
    v2h hv;
    hv.x = (_Float16)(val.x * sS[c]);
    hv.y = (_Float16)(val.y * sS[c + 1]);
    *(v2h*)(xs + r * ROWS + p * PXS + c) = hv;
  }
  __syncthreads();

  // ---- per-wave tiling: 4(M) x 2(N) waves; each wave 32px x 64ch = 2x4 C-tiles ----
  const int lane = tid & 31;
  const int wv   = tid >> 5;
  const int lm   = lane & 15;
  const int hs   = lane >> 4;          // half-select
  const int mT   = (wv & 3) * 32;      // pixel base of wave
  const int nT   = (wv >> 2) * 64;     // out-channel base of wave

  v8f acc[2][4];
#pragma unroll
  for (int mi = 0; mi < 2; ++mi)
#pragma unroll
    for (int ni = 0; ni < 4; ++ni)
      acc[mi][ni] = (v8f){0.f, 0.f, 0.f, 0.f, 0.f, 0.f, 0.f, 0.f};

  for (int kh = 0; kh < 3; ++kh) {
#pragma unroll
    for (int kw = 0; kw < 3; ++kw) {
      const _Float16* wbase = wT + (size_t)(kh * 3 + kw) * (Cin * Cout);
#pragma unroll
      for (int kc = 0; kc < 4; ++kc) {
        const int k0 = kc * 32;
        // A fragments: 16x32 f16, ISA layout (lanes<16: K{0..7,16..23}; lanes>=16: K{8..15,24..31})
        v16h a[2];
#pragma unroll
        for (int mi = 0; mi < 2; ++mi) {
          const int m = mT + mi * 16 + lm;
          const _Float16* ap = xs + kh * ROWS + (m + kw) * PXS + k0 + hs * 8;
          v8h lo = *(const v8h*)ap;
          v8h hi = *(const v8h*)(ap + 16);
          a[mi] = __builtin_shufflevector(lo, hi, 0, 1, 2, 3, 4, 5, 6, 7,
                                                  8, 9, 10, 11, 12, 13, 14, 15);
        }
        // B fragments: 32x16 f16, lane=column, same K layout as A (K{0..7,16..23} / K{8..15,24..31}) from wT[t][o][i]
        v16h b[4];
#pragma unroll
        for (int ni = 0; ni < 4; ++ni) {
          const int o = nT + ni * 16 + lm;
          const _Float16* bp = wbase + (size_t)o * Cin + k0 + hs * 8;
          v8h lo = *(const v8h*)bp;
          v8h hi = *(const v8h*)(bp + 16);
          b[ni] = __builtin_shufflevector(lo, hi, 0, 1, 2, 3, 4, 5, 6, 7,
                                                  8, 9, 10, 11, 12, 13, 14, 15);
        }
#pragma unroll
        for (int mi = 0; mi < 2; ++mi)
#pragma unroll
          for (int ni = 0; ni < 4; ++ni)
            acc[mi][ni] = __builtin_amdgcn_wmma_f32_16x16x32_f16(
                false, a[mi], false, b[ni], (short)0, acc[mi][ni], false, false);
      }
    }
  }

  // ---- fused epilogue: demod, noise, bias, leaky-relu * sqrt(2) ----
  const float ns = nsPtr[0];
#pragma unroll
  for (int mi = 0; mi < 2; ++mi) {
#pragma unroll
    for (int ni = 0; ni < 4; ++ni) {
      const int o = nT + ni * 16 + lm;
      const float dv = dS[o];
      const float bv = bS[o];
      v8f cc = acc[mi][ni];
#pragma unroll
      for (int r = 0; r < 8; ++r) {
        const int m = mT + mi * 16 + r + hs * 8;   // C layout: VGPR r -> M=r (lanes<16) / r+8
        const int w = w0 + m;
        const float nz = noise[((size_t)n * Hdim + h) * Wdim + w];
        float y = cc[r] * dv + ns * nz + bv;
        y = (y > 0.f ? y : 0.2f * y) * 1.41421356237f;
        out[(((size_t)n * Hdim + h) * Wdim + w) * Cout + o] = y;
      }
    }
  }
}

extern "C" void kernel_launch(void* const* d_in, const int* in_sizes, int n_in,
                              void* d_out, int out_size, void* d_ws, size_t ws_size,
                              hipStream_t stream) {
  const float* x    = (const float*)d_in[0];
  const float* dlat = (const float*)d_in[1];
  const float* noise= (const float*)d_in[2];
  const float* aw   = (const float*)d_in[3];
  const float* ab   = (const float*)d_in[4];
  const float* cw   = (const float*)d_in[5];
  const float* cb   = (const float*)d_in[6];
  const float* ns   = (const float*)d_in[7];
  const int*   li   = (const int*)d_in[8];
  float* out = (float*)d_out;

  char* ws = (char*)d_ws;
  float*     sBuf = (float*)ws;                 // 8*128 f32
  float*     dBuf = (float*)(ws + 4096);        // 8*128 f32
  _Float16*  wTb  = (_Float16*)(ws + 8192);     // 9*128*128 f16

  prep_affine<<<dim3(8), 128, 0, stream>>>(dlat, aw, ab, li, sBuf);
  prep_demod<<<dim3(8), 128, 0, stream>>>(cw, sBuf, dBuf);
  prep_weights<<<dim3(9 * 128 * 128 / 256), 256, 0, stream>>>(cw, wTb);

  const size_t smem = 3 * Cin * sizeof(float) + XS_HALFS * sizeof(_Float16);
  synth_main<<<dim3(2, Hdim, 8), 256, smem, stream>>>(x, noise, sBuf, dBuf, wTb,
                                                      cb, ns, out);
}
